// TransformerModel_29987461661391
// MI455X (gfx1250) — matrix operand order repaired and checked by reading, not compiled, not run on hardware
//
#include <hip/hip_runtime.h>
#include <stdint.h>
#include <stddef.h>

// ---------------- types ----------------
typedef __attribute__((ext_vector_type(16))) __bf16 v16bf;
typedef __attribute__((ext_vector_type(8)))  float  v8f;

struct U4 { unsigned x, y, z, w; };

union Frag32B {
  U4 u[2];
  v16bf v;
  unsigned short h[16];
};
static_assert(sizeof(Frag32B) == 32, "frag size");

__device__ __forceinline__ unsigned short f2bf(float f) {
  union { float f; unsigned u; } x; x.f = f;
  unsigned r = x.u + 0x7FFFu + ((x.u >> 16) & 1u);  // round-to-nearest-even
  return (unsigned short)(r >> 16);
}

// async 16B global->LDS copy (ASYNCcnt-tracked)
__device__ __forceinline__ void async_b128(unsigned lds_byte_addr, const void* gaddr) {
  asm volatile("global_load_async_to_lds_b128 %0, %1, off"
               :: "v"(lds_byte_addr), "v"((unsigned long long)(size_t)gaddr)
               : "memory");
}
// wait for all async ops
__device__ __forceinline__ void wait_async_le0() {
  asm volatile("s_wait_asynccnt 0" ::: "memory");
}
// wait until only the newest stage (4 wave-wide async ops) remains outstanding
__device__ __forceinline__ void wait_async_le4() {
  asm volatile("s_wait_asynccnt 4" ::: "memory");
}

// model dims
#define NB     8
#define NPAT   1024
#define ROWS   8192      // NB*NPAT
#define FDIM   360
#define FPAD   384
#define QKVW   1080
#define QKVPAD 1152
#define HEADS  8
#define HD     45
#define HDP    64        // padded per-head dim for QK^T K-dimension
#define HDO    48        // padded per-head dim for PV N-dimension
#define NLAYER 4

// ---------------- weight/bias padding kernels ----------------
// transpose+pad+convert: src[K,N] f32 -> dst[Npad,Kpad] bf16 ([n][k])
__global__ void cvt_pad_bf16_T(const float* __restrict__ s, unsigned short* __restrict__ d,
                               int K, int N, int ldn, int Kpad, int Npad) {
  int t = blockIdx.x * blockDim.x + threadIdx.x;
  if (t >= Npad * Kpad) return;
  int n = t / Kpad, k = t % Kpad;
  float v = (k < K && n < N) ? s[(size_t)k * ldn + n] : 0.f;
  d[t] = f2bf(v);
}

__global__ void pad_f32(const float* __restrict__ s, float* __restrict__ d,
                        int rs, int cs, int lds_, int rd, int ldd) {
  int t = blockIdx.x * blockDim.x + threadIdx.x;
  if (t >= rd * ldd) return;
  int r = t / ldd, c = t % ldd;
  d[t] = (r < rs && c < cs) ? s[(size_t)r * lds_ + c] : 0.f;
}

// ---------------- patchify: images -> bf16 patch matrix [8192,256] ----------------
__global__ void patchify(const float* __restrict__ img, unsigned short* __restrict__ Ap) {
  int t = blockIdx.x * 256 + threadIdx.x;
  if (t >= ROWS * 256) return;
  int c = t & 255, r = t >> 8;
  int b = r >> 10, p = r & 1023;
  int py = p >> 5, px = p & 31;
  int iy = c >> 4, ix = c & 15;
  float v = img[((size_t)b * 512 + py * 16 + iy) * 512 + px * 16 + ix];
  Ap[t] = f2bf(v);
}

// ---------------- generic bf16 WMMA GEMM ----------------
// C[M,N](f32) = A[M,K](bf16,row-major) * Bt[N,K](bf16, pre-transposed) + bias[n] (+pos)
// block = 128 threads (4 waves), 64x64 tile, k-step 32,
// DOUBLE-BUFFERED async-to-LDS staging overlapped with WMMA via partial ASYNCcnt wait.
__global__ __launch_bounds__(128) void gemm_bf16(
    const unsigned short* __restrict__ A, int lda,
    const unsigned short* __restrict__ Bt, int ldb,   // ldb = Kpad
    float* __restrict__ C, int ldc,
    int K,
    const float* __restrict__ bias,
    const float* __restrict__ pos, int pos_ld) {
  __shared__ __align__(16) unsigned short sA[2][64 * 32];   // [buf][m][k]
  __shared__ __align__(16) unsigned short sB[2][64 * 32];   // [buf][n][k]
  const int tid = threadIdx.x;
  const int lane = tid & 31, w = tid >> 5;
  const int lane16 = lane & 15, hi = lane >> 4;
  const int m0 = blockIdx.y * 64, n0 = blockIdx.x * 64;
  const unsigned sAb = (unsigned)(size_t)(void*)&sA[0][0];
  const unsigned sBb = (unsigned)(size_t)(void*)&sB[0][0];

  // issue one stage (64x32 A tile + 64x32 B tile) as 4 wave-wide async b128 ops
  auto stage = [&](int k0, int buf) {
#pragma unroll
    for (int i = 0; i < 2; ++i) {
      int ch = tid + 128 * i;            // 0..255, byte addr = ch*16
      int r = ch >> 2, cc = ch & 3;
      async_b128(sAb + (unsigned)(buf * 4096 + ch * 16),
                 &A[(size_t)(m0 + r) * lda + k0 + cc * 8]);
      async_b128(sBb + (unsigned)(buf * 4096 + ch * 16),
                 &Bt[(size_t)(n0 + r) * ldb + k0 + cc * 8]);
    }
  };

  v8f acc[4];
  const v8f vz = {0.f, 0.f, 0.f, 0.f, 0.f, 0.f, 0.f, 0.f};
#pragma unroll
  for (int c = 0; c < 4; ++c) acc[c] = vz;

  stage(0, 0);
  int buf = 0;
  for (int k0 = 0; k0 < K; k0 += 32) {
    if (k0 + 32 < K) {
      stage(k0 + 32, buf ^ 1);   // prefetch next stage into the other buffer
      wait_async_le4();          // require only current stage complete
    } else {
      wait_async_le0();
    }
    __syncthreads();

    const unsigned short* pA = sA[buf];
    const unsigned short* pB = sB[buf];
    // A fragment: lane holds M=lane16 of row-subtile w; K halves 16hi..16hi+15
    Frag32B a;
    a.u[0] = *(const U4*)&pA[(16 * w + lane16) * 32 + 16 * hi];
    a.u[1] = *(const U4*)&pA[(16 * w + lane16) * 32 + 16 * hi + 8];
#pragma unroll
    for (int c = 0; c < 4; ++c) {
      // B fragment: lane holds N=16c+lane16; K halves 16hi..16hi+15
      Frag32B bf;
      bf.u[0] = *(const U4*)&pB[(16 * c + lane16) * 32 + 16 * hi];
      bf.u[1] = *(const U4*)&pB[(16 * c + lane16) * 32 + 16 * hi + 8];
      acc[c] = __builtin_amdgcn_wmma_f32_16x16x32_bf16(
          false, a.v, false, bf.v, (short)0, acc[c], false, false);
    }
    __syncthreads();             // all waves done reading buf before it is restaged
    buf ^= 1;
  }

#pragma unroll
  for (int c = 0; c < 4; ++c) {
#pragma unroll
    for (int j = 0; j < 8; ++j) {
      int m = m0 + 16 * w + j + 8 * hi;
      int n = n0 + 16 * c + lane16;
      float v = acc[c][j] + bias[n];
      if (pos) v += pos[(size_t)(m % NPAT) * pos_ld + n];
      C[(size_t)m * ldc + n] = v;
    }
  }
}

// ---------------- layernorm: f32 X[8192,384] -> bf16 Z[8192,384] (pads zero) ----------------
__global__ __launch_bounds__(128) void ln_kernel(const float* __restrict__ X,
                                                 const float* __restrict__ g,
                                                 const float* __restrict__ bta,
                                                 unsigned short* __restrict__ Z) {
  int lane = threadIdx.x & 31, w = threadIdx.x >> 5;
  int row = blockIdx.x * 4 + w;
  const float* xr = &X[(size_t)row * FPAD];
  float v[12], s = 0.f, sq = 0.f;
#pragma unroll
  for (int i = 0; i < 12; ++i) {
    int c = lane + 32 * i;
    float x = (c < FDIM) ? xr[c] : 0.f;
    v[i] = x; s += x; sq += x * x;
  }
#pragma unroll
  for (int m = 1; m < 32; m <<= 1) {
    s += __shfl_xor(s, m, 32);
    sq += __shfl_xor(sq, m, 32);
  }
  float mu = s * (1.f / FDIM);
  float var = sq * (1.f / FDIM) - mu * mu;
  float inv = rsqrtf(var + 1e-6f);
#pragma unroll
  for (int i = 0; i < 12; ++i) {
    int c = lane + 32 * i;
    float z = (c < FDIM) ? (v[i] - mu) * inv * g[c] + bta[c] : 0.f;
    Z[(size_t)row * FPAD + c] = f2bf(z);
  }
}

// ---------------- QKV packing for WMMA-friendly attention layouts ----------------
// Qb/Kb: [64 bh][1024 tok][64 d] bf16 (d padded, Q pre-scaled by 1/sqrt(D))
__global__ void pack_qk(const float* __restrict__ qkv, unsigned short* __restrict__ Qb,
                        unsigned short* __restrict__ Kb) {
  int t = blockIdx.x * 256 + threadIdx.x;
  if (t >= 64 * NPAT * HDP) return;
  int d = t & 63, tok = (t >> 6) & 1023, bh = t >> 16;
  int b = bh >> 3, h = bh & 7;
  const float scale = rsqrtf((float)HD);
  size_t base = (size_t)(b * NPAT + tok) * QKVPAD + h * HD;
  float q = (d < HD) ? qkv[base + d] * scale : 0.f;
  float k = (d < HD) ? qkv[base + FDIM + d] : 0.f;
  Qb[t] = f2bf(q);
  Kb[t] = f2bf(k);
}

// Vt: [64 bh][48 d][1024 key] bf16 (transposed so PV B-fragments are contiguous)
__global__ void pack_v(const float* __restrict__ qkv, unsigned short* __restrict__ Vt) {
  int t = blockIdx.x * 256 + threadIdx.x;
  if (t >= 64 * HDO * NPAT) return;
  int key = t & 1023, d = (t >> 10) % HDO, bh = t / (HDO * NPAT);
  int b = bh >> 3, h = bh & 7;
  float v = (d < HD) ? qkv[(size_t)(b * NPAT + key) * QKVPAD + 2 * FDIM + h * HD + d] : 0.f;
  Vt[t] = f2bf(v);
}

// ---------------- fused flash attention ----------------
// one wave per (b,h, 16-query block); packed bf16 Q/K/V; O bf16 [8192,384].
__global__ __launch_bounds__(128) void attn_kernel(const unsigned short* __restrict__ Qb,
                                                   const unsigned short* __restrict__ Kb,
                                                   const unsigned short* __restrict__ Vt,
                                                   unsigned short* __restrict__ O) {
  __shared__ __align__(16) unsigned short pb[4 * 16 * 32];  // per-wave P transpose buffer
  const int lane = threadIdx.x & 31, w = threadIdx.x >> 5;
  const int lane16 = lane & 15, hi = lane >> 4;
  unsigned short* pbuf = &pb[w * 16 * 32];

  int gw = blockIdx.x * 4 + w;                 // 0..4095
  int bh = gw >> 6;                            // (b*8+h)
  int q0 = (gw & 63) * 16;
  int b = bh >> 3, h = bh & 7;

  const unsigned short* Qbh = Qb + (size_t)bh * NPAT * HDP;
  const unsigned short* Kbh = Kb + (size_t)bh * NPAT * HDP;
  const unsigned short* Vbh = Vt + (size_t)bh * HDO * NPAT;

  // Q fragments (A layout): K halves 16hi..16hi+15 within each 32-d step
  Frag32B qf[2];
#pragma unroll
  for (int ks = 0; ks < 2; ++ks) {
    const unsigned short* qr = &Qbh[(size_t)(q0 + lane16) * HDP + 32 * ks];
    qf[ks].u[0] = *(const U4*)&qr[16 * hi];
    qf[ks].u[1] = *(const U4*)&qr[16 * hi + 8];
  }

  float m_run[8], s_run[8];
  v8f accO[3];
  const v8f vz = {0.f, 0.f, 0.f, 0.f, 0.f, 0.f, 0.f, 0.f};
#pragma unroll
  for (int j = 0; j < 8; ++j) { m_run[j] = -1e30f; s_run[j] = 0.f; }
#pragma unroll
  for (int c = 0; c < 3; ++c) accO[c] = vz;

  for (int kb = 0; kb < NPAT / 32; ++kb) {
    int key0 = kb * 32;
    if (kb + 1 < NPAT / 32) {  // prefetch next key block (K rows + V columns)
      __builtin_prefetch(&Kbh[(size_t)(key0 + 32 + lane16) * HDP], 0, 0);
      __builtin_prefetch(&Vbh[(size_t)lane16 * NPAT + key0 + 32], 0, 0);
    }
    // ---- S = Q*K^T for two 16-key tiles ----
    v8f st[2];
    st[0] = vz; st[1] = vz;
#pragma unroll
    for (int t = 0; t < 2; ++t) {
      const unsigned short* kr = &Kbh[(size_t)(key0 + 16 * t + lane16) * HDP];
#pragma unroll
      for (int ks = 0; ks < 2; ++ks) {
        Frag32B kf;  // B layout: K halves 32ks + 16hi .. +15 (contiguous)
        kf.u[0] = *(const U4*)&kr[32 * ks + 16 * hi];
        kf.u[1] = *(const U4*)&kr[32 * ks + 16 * hi + 8];
        st[t] = __builtin_amdgcn_wmma_f32_16x16x32_bf16(
            false, qf[ks].v, false, kf.v, (short)0, st[t], false, false);
      }
    }
    // ---- online softmax over the 32 keys ----
#pragma unroll
    for (int j = 0; j < 8; ++j) {
      float mx = fmaxf(st[0][j], st[1][j]);
#pragma unroll
      for (int m = 1; m < 16; m <<= 1) mx = fmaxf(mx, __shfl_xor(mx, m, 32));
      float mnew = fmaxf(m_run[j], mx);
      float corr = __expf(m_run[j] - mnew);
      float p0 = __expf(st[0][j] - mnew);
      float p1 = __expf(st[1][j] - mnew);
      float rs = p0 + p1;
#pragma unroll
      for (int m = 1; m < 16; m <<= 1) rs += __shfl_xor(rs, m, 32);
      s_run[j] = s_run[j] * corr + rs;
      m_run[j] = mnew;
#pragma unroll
      for (int c = 0; c < 3; ++c) accO[c][j] *= corr;
      int r = j + 8 * hi;                       // C layout row
      pbuf[r * 32 + lane16] = f2bf(p0);
      pbuf[r * 32 + 16 + lane16] = f2bf(p1);
    }
    asm volatile("s_wait_dscnt 0" ::: "memory"); // LDS write->read within wave
    // ---- O += P*V ----
    Frag32B pf;
    pf.u[0] = *(const U4*)&pbuf[lane16 * 32 + 16 * hi];
    pf.u[1] = *(const U4*)&pbuf[lane16 * 32 + 16 * hi + 8];
#pragma unroll
    for (int c = 0; c < 3; ++c) {
      const unsigned short* vr = &Vbh[(size_t)(16 * c + lane16) * NPAT + key0 + 16 * hi];
      Frag32B vf;  // B layout: K halves = 16 contiguous keys
      vf.u[0] = *(const U4*)&vr[0];
      vf.u[1] = *(const U4*)&vr[8];
      accO[c] = __builtin_amdgcn_wmma_f32_16x16x32_bf16(
          false, pf.v, false, vf.v, (short)0, accO[c], false, false);
    }
  }

  // normalize + write bf16 O (valid head cols only; pads pre-zeroed)
#pragma unroll
  for (int j = 0; j < 8; ++j) {
    float inv = 1.f / s_run[j];
    int row = b * NPAT + q0 + j + 8 * hi;
#pragma unroll
    for (int c = 0; c < 3; ++c) {
      int d = 16 * c + lane16;
      if (d < HD) O[(size_t)row * FPAD + h * HD + d] = f2bf(accO[c][j] * inv);
    }
  }
}

// ---------------- fused conv head: 5x5 conv(32ch) + ReLU + 1x1 conv ----------------
__global__ void conv_head(const float* __restrict__ X, const float* __restrict__ k1,
                          const float* __restrict__ b1, const float* __restrict__ k2,
                          const float* __restrict__ b2, float* __restrict__ out) {
  int t = blockIdx.x * 256 + threadIdx.x;
  if (t >= NB * NPAT * FDIM) return;
  int f = t % FDIM;
  int n = (t / FDIM) % NPAT;
  int b = t / (FDIM * NPAT);
  float a[32];
#pragma unroll
  for (int c = 0; c < 32; ++c) a[c] = b1[c];
  for (int dy = 0; dy < 5; ++dy) {
    int nn = n + dy - 2;
    if (nn < 0 || nn >= NPAT) continue;
    for (int dx = 0; dx < 5; ++dx) {
      int ff = f + dx - 2;
      if (ff < 0 || ff >= FDIM) continue;
      float xv = X[(size_t)(b * NPAT + nn) * FPAD + ff];
      const float* kr = &k1[(dy * 5 + dx) * 32];
#pragma unroll
      for (int c = 0; c < 32; ++c) a[c] += xv * kr[c];
    }
  }
  float acc = b2[0];
#pragma unroll
  for (int c = 0; c < 32; ++c) acc += fmaxf(a[c], 0.f) * k2[c];
  out[t] = acc;
}

// ---------------- host orchestration ----------------
static inline int cdiv(int a, int b) { return (a + b - 1) / b; }

extern "C" void kernel_launch(void* const* d_in, const int* in_sizes, int n_in,
                              void* d_out, int out_size, void* d_ws, size_t ws_size,
                              hipStream_t stream) {
  (void)in_sizes; (void)n_in; (void)out_size; (void)ws_size;
  const float* images = (const float*)d_in[0];
  const float* Wp     = (const float*)d_in[1];
  const float* bp     = (const float*)d_in[2];
  const float* pos    = (const float*)d_in[3];
  const float* ln_g   = (const float*)d_in[4];
  const float* ln_b   = (const float*)d_in[5];
  const float* Wqkv   = (const float*)d_in[6];
  const float* bqkv   = (const float*)d_in[7];
  const float* Wproj  = (const float*)d_in[8];
  const float* bproj  = (const float*)d_in[9];
  const float* c1k    = (const float*)d_in[10];
  const float* c1b    = (const float*)d_in[11];
  const float* c2k    = (const float*)d_in[12];
  const float* c2b    = (const float*)d_in[13];
  float* out = (float*)d_out;

  char* ws = (char*)d_ws;
  size_t off = 0;
  auto carve = [&](size_t bytes) { void* p = ws + off; off += (bytes + 255) & ~(size_t)255; return p; };

  // weights stored TRANSPOSED [N][K] for contiguous B-tile staging
  unsigned short* WpB    = (unsigned short*)carve((size_t)FPAD * 256 * 2);
  unsigned short* WqkvB  = (unsigned short*)carve((size_t)NLAYER * QKVPAD * FPAD * 2);
  unsigned short* WprojB = (unsigned short*)carve((size_t)NLAYER * FPAD * FPAD * 2);
  float* bpP   = (float*)carve(FPAD * 4);
  float* bqkvP = (float*)carve((size_t)NLAYER * QKVPAD * 4);
  float* bprojP= (float*)carve((size_t)NLAYER * FPAD * 4);
  float* posP  = (float*)carve((size_t)NPAT * FPAD * 4);
  unsigned short* Ap = (unsigned short*)carve((size_t)ROWS * 256 * 2);
  float* X   = (float*)carve((size_t)ROWS * FPAD * 4);
  unsigned short* Z  = (unsigned short*)carve((size_t)ROWS * FPAD * 2);
  float* QKV = (float*)carve((size_t)ROWS * QKVPAD * 4);
  unsigned short* Obuf = (unsigned short*)carve((size_t)ROWS * FPAD * 2);
  unsigned short* Qb = (unsigned short*)carve((size_t)64 * NPAT * HDP * 2);
  unsigned short* Kb = (unsigned short*)carve((size_t)64 * NPAT * HDP * 2);
  unsigned short* Vt = (unsigned short*)carve((size_t)64 * HDO * NPAT * 2);

  // --- weight/bias padding (transposed) ---
  cvt_pad_bf16_T<<<cdiv(FPAD * 256, 256), 256, 0, stream>>>(Wp, WpB, 256, FDIM, FDIM, 256, FPAD);
  for (int l = 0; l < NLAYER; ++l) {
    cvt_pad_bf16_T<<<cdiv(QKVPAD * FPAD, 256), 256, 0, stream>>>(
        Wqkv + (size_t)l * FDIM * QKVW, WqkvB + (size_t)l * QKVPAD * FPAD,
        FDIM, QKVW, QKVW, FPAD, QKVPAD);
    cvt_pad_bf16_T<<<cdiv(FPAD * FPAD, 256), 256, 0, stream>>>(
        Wproj + (size_t)l * FDIM * FDIM, WprojB + (size_t)l * FPAD * FPAD,
        FDIM, FDIM, FDIM, FPAD, FPAD);
  }
  pad_f32<<<cdiv(FPAD, 256), 256, 0, stream>>>(bp, bpP, 1, FDIM, FDIM, 1, FPAD);
  pad_f32<<<cdiv(NLAYER * QKVPAD, 256), 256, 0, stream>>>(bqkv, bqkvP, NLAYER, QKVW, QKVW, NLAYER, QKVPAD);
  pad_f32<<<cdiv(NLAYER * FPAD, 256), 256, 0, stream>>>(bproj, bprojP, NLAYER, FDIM, FDIM, NLAYER, FPAD);
  pad_f32<<<cdiv(NPAT * FPAD, 256), 256, 0, stream>>>(pos, posP, NPAT, FDIM, FDIM, NPAT, FPAD);
  hipMemsetAsync(Obuf, 0, (size_t)ROWS * FPAD * 2, stream);  // zero pads of attention output

  // --- patch encoder ---
  patchify<<<ROWS, 256, 0, stream>>>(images, Ap);
  gemm_bf16<<<dim3(FPAD / 64, ROWS / 64), 128, 0, stream>>>(
      Ap, 256, WpB, 256, X, FPAD, 256, bpP, posP, FPAD);

  // --- transformer layers ---
  for (int l = 0; l < NLAYER; ++l) {
    ln_kernel<<<ROWS / 4, 128, 0, stream>>>(X, ln_g + (size_t)l * FDIM, ln_b + (size_t)l * FDIM, Z);
    gemm_bf16<<<dim3(QKVPAD / 64, ROWS / 64), 128, 0, stream>>>(
        Z, FPAD, WqkvB + (size_t)l * QKVPAD * FPAD, FPAD, QKV, QKVPAD, FPAD,
        bqkvP + (size_t)l * QKVPAD, nullptr, 0);
    pack_qk<<<cdiv(64 * NPAT * HDP, 256), 256, 0, stream>>>(QKV, Qb, Kb);
    pack_v<<<cdiv(64 * HDO * NPAT, 256), 256, 0, stream>>>(QKV, Vt);
    attn_kernel<<<NB * HEADS * 64 / 4, 128, 0, stream>>>(Qb, Kb, Vt, Obuf);
    gemm_bf16<<<dim3(FPAD / 64, ROWS / 64), 128, 0, stream>>>(
        Obuf, FPAD, WprojB + (size_t)l * FPAD * FPAD, FPAD, X, FPAD, FPAD,
        bprojP + (size_t)l * FPAD, nullptr, 0);
  }

  // --- conv head ---
  conv_head<<<cdiv(NB * NPAT * FDIM, 256), 256, 0, stream>>>(X, c1k, c1b, c2k, c2b, out);
}
